// GPT_45586782880069
// MI455X (gfx1250) — compile-verified, matrix operand order checked by reading, not run on hardware
//
#include <hip/hip_runtime.h>
#include <cstdint>

#define D_MODEL 1024
#define N_HEADS 16
#define HEAD_DIM 64
#define N_LAYERS 6
#define VOCAB 50257
#define SEQ_B 2
#define SEQ_T 1024
#define ROWS (SEQ_B * SEQ_T)

typedef __bf16 bf16;
typedef __attribute__((ext_vector_type(16))) __bf16 v16bf;
typedef __attribute__((ext_vector_type(8))) float v8f;
typedef __attribute__((ext_vector_type(4))) int v4i;
typedef __attribute__((ext_vector_type(8))) int v8i;

__device__ __forceinline__ void wave_lds_fence() {
  // CDNA5 split-counter wait: order LDS writes before cross-lane LDS reads
  asm volatile("s_wait_dscnt 0x0" ::: "memory");
}

__device__ __forceinline__ v8f zero8() {
  v8f z;
#pragma unroll
  for (int i = 0; i < 8; ++i) z[i] = 0.f;
  return z;
}

// ---------------------------------------------------------------------------
// Tensor Data Mover: 2D tile load (rows x row_bytes, byte-granular) into LDS.
// D# layout per CDNA5 ISA 8.3/8.4: group0 = {count, lds_addr, gaddr, type=2},
// group1 = {data_size=1B, tensor dims (set huge: no OOB), tile dims, stride}.
// Issued by one wave; completion via s_wait_tensorcnt.
// ---------------------------------------------------------------------------
__device__ __forceinline__ void tdm_load_2d(uint32_t lds_off, const void* gaddr,
                                            uint32_t row_bytes, uint32_t rows,
                                            uint32_t stride_bytes) {
  const uint64_t ga = (uint64_t)gaddr;
  v4i g0;
  g0[0] = 1;                                                // count=1
  g0[1] = (int)lds_off;                                     // lds_addr
  g0[2] = (int)(uint32_t)ga;                                // global_addr[31:0]
  g0[3] = (int)((uint32_t)((ga >> 32) & 0x1FFFFFFu) | (2u << 30)); // [56:32]|type=2
  v8i g1;
  g1[0] = 0;                                  // wg_mask=0, data_size=0 (1 byte)
  g1[1] = (int)(0x7FFFu << 16);               // tensor_dim0[15:0] (huge)
  g1[2] = (int)(0x0000FFFFu | (0xFFFFu << 16));  // tdim0 hi | tensor_dim1 lo
  g1[3] = (int)(0x0000FFFFu | (row_bytes << 16));// tdim1 hi | tile_dim0
  g1[4] = (int)(rows & 0xFFFFu);              // tile_dim1 | tile_dim2=0
  g1[5] = (int)stride_bytes;                  // tensor_dim0_stride[31:0]
  g1[6] = 0;                                  // stride hi | dim1_stride lo
  g1[7] = 0;
  asm volatile("tensor_load_to_lds %0, %1" ::"s"(g0), "s"(g1) : "memory");
}

// 16-bit A/B fragment from a row-major 32-wide slice (ISA 7.12.2 16-bit 16x32):
// lane (idx = lane&15) selects the row; half = lane>>4.
// elements j: K = half*8+j ; elements 8+j: K = 16 + half*8 + j
__device__ __forceinline__ v16bf frag_row(const bf16* __restrict__ row, int half) {
  v16bf f;
#pragma unroll
  for (int j = 0; j < 8; ++j) {
    f[j]     = row[half * 8 + j];
    f[8 + j] = row[16 + half * 8 + j];
  }
  return f;
}

// ---------------------------------------------------------------------------
// Embedding: h[row][c] = wte[x[row]][c] + wpe[row % T][c]
// ---------------------------------------------------------------------------
__global__ __launch_bounds__(256) void embed_kernel(const int* __restrict__ x,
                                                    const float* __restrict__ wte,
                                                    const float* __restrict__ wpe,
                                                    float* __restrict__ h) {
  const int row = blockIdx.x;
  const int t = row & (SEQ_T - 1);
  const long tok = x[row];
  const float* te = wte + (size_t)tok * D_MODEL;
  const float* pe = wpe + (size_t)t * D_MODEL;
  float* out = h + (size_t)row * D_MODEL;
#pragma unroll
  for (int j = 0; j < 4; ++j) {
    const int c = threadIdx.x + j * 256;
    out[c] = te[c] + pe[c];
  }
}

// ---------------------------------------------------------------------------
// LayerNorm (fp32 in) -> bf16 out, one workgroup per row
// ---------------------------------------------------------------------------
__global__ __launch_bounds__(256) void ln_kernel(const float* __restrict__ h,
                                                 const float* __restrict__ g,
                                                 const float* __restrict__ b,
                                                 bf16* __restrict__ out) {
  __shared__ float r1[256];
  __shared__ float r2[256];
  const int row = blockIdx.x;
  const float* xr = h + (size_t)row * D_MODEL;
  float s = 0.f, s2 = 0.f;
#pragma unroll
  for (int j = 0; j < 4; ++j) {
    const float v = xr[threadIdx.x + j * 256];
    s += v;
    s2 += v * v;
  }
  r1[threadIdx.x] = s;
  r2[threadIdx.x] = s2;
  __syncthreads();
  for (int off = 128; off > 0; off >>= 1) {
    if (threadIdx.x < off) {
      r1[threadIdx.x] += r1[threadIdx.x + off];
      r2[threadIdx.x] += r2[threadIdx.x + off];
    }
    __syncthreads();
  }
  const float mu = r1[0] * (1.f / D_MODEL);
  const float var = r2[0] * (1.f / D_MODEL) - mu * mu;
  const float rstd = rsqrtf(var + 1e-5f);
  bf16* orow = out + (size_t)row * D_MODEL;
#pragma unroll
  for (int j = 0; j < 4; ++j) {
    const int c = threadIdx.x + j * 256;
    orow[c] = (bf16)((xr[c] - mu) * rstd * g[c] + b[c]);
  }
}

// ---------------------------------------------------------------------------
// WMMA GEMM: out = act(A[M,K](bf16) @ W + bias)
//   TRANSB=0: W fp32 [K,N]    TRANSB=1: W fp32 [N,K] (lm_head)
//   RESID=1 : Cf[m,n] += v   (residual accumulate into fp32 h)
//   OUTBF16 : write bf16     else fp32
// 128x128 macro tile, 8 waves (2x4), each wave 64x32 (4x2 WMMA tiles), BK=32.
// A tiles double-buffered via Tensor Data Mover (wave 0 issues, overlap with
// WMMA); B tiles cooperative-loaded (fp32 -> bf16 conversion on the fly) with
// global_prefetch of the next tile.
// ---------------------------------------------------------------------------
template <int GELU, int RESID, int OUTBF16, int TRANSB>
__global__ __launch_bounds__(256) void gemm_kernel(const bf16* __restrict__ A,
                                                   const float* __restrict__ W,
                                                   const float* __restrict__ bias,
                                                   float* __restrict__ Cf,
                                                   bf16* __restrict__ Cb,
                                                   int M, int N, int K) {
  __shared__ bf16 As[2][128][32];
  __shared__ bf16 Bs[128][32];  // transposed: [n][k]

  const int tid = threadIdx.x;
  const int lane = tid & 31;
  const int wave = tid >> 5;
  const int wm = wave >> 2;  // 0..1
  const int wn = wave & 3;   // 0..3
  const int half = lane >> 4;
  const int l16 = lane & 15;
  const int m0 = blockIdx.y * 128;
  const int n0 = blockIdx.x * 128;

  v8f acc[4][2];
#pragma unroll
  for (int mt = 0; mt < 4; ++mt)
#pragma unroll
    for (int nt = 0; nt < 2; ++nt) acc[mt][nt] = zero8();

  // prologue: TDM fetch of the first A tile (128 rows x 64 bytes)
  if (wave == 0) {
    tdm_load_2d((uint32_t)(size_t)&As[0][0][0], &A[(size_t)m0 * K], 64u, 128u,
                (uint32_t)K * 2u);
  }

  int buf = 0;
  for (int k0 = 0; k0 < K; k0 += 32, buf ^= 1) {
    // ---- B tile (fp32 -> bf16, stored transposed Bs[n][k]) ----
    if (TRANSB) {
      if (k0 + 32 < K && tid < 128 && n0 + tid < N)
        __builtin_prefetch(&W[(size_t)(n0 + tid) * K + k0 + 32], 0, 1);
#pragma unroll
      for (int i = tid; i < 1024; i += 256) {
        const int n = i >> 3;
        const int c = (i & 7) << 2;
        float4 f = make_float4(0.f, 0.f, 0.f, 0.f);
        if (n0 + n < N) f = *(const float4*)&W[(size_t)(n0 + n) * K + k0 + c];
        Bs[n][c + 0] = (bf16)f.x;
        Bs[n][c + 1] = (bf16)f.y;
        Bs[n][c + 2] = (bf16)f.z;
        Bs[n][c + 3] = (bf16)f.w;
      }
    } else {
      if (k0 + 32 < K)
        __builtin_prefetch(&W[(size_t)(k0 + 32 + (tid >> 3)) * N + n0 +
                              ((tid & 7) << 4)], 0, 1);
#pragma unroll
      for (int i = tid; i < 1024; i += 256) {
        const int k = i >> 5;
        const int n = (i & 31) << 2;
        const float4 f = *(const float4*)&W[(size_t)(k0 + k) * N + n0 + n];
        Bs[n + 0][k] = (bf16)f.x;
        Bs[n + 1][k] = (bf16)f.y;
        Bs[n + 2][k] = (bf16)f.z;
        Bs[n + 3][k] = (bf16)f.w;
      }
    }
    // A tile for this step is TDM-in-flight: issuing wave drains TENSORcnt,
    // then the workgroup barrier publishes both As[buf] and Bs.
    if (wave == 0) __builtin_amdgcn_s_wait_tensorcnt(0);
    __syncthreads();

    // kick off the DMA of the NEXT A tile into the other buffer; it overlaps
    // with the WMMA work below.
    if (wave == 0 && k0 + 32 < K) {
      tdm_load_2d((uint32_t)(size_t)&As[buf ^ 1][0][0],
                  &A[(size_t)m0 * K + (k0 + 32)], 64u, 128u, (uint32_t)K * 2u);
    }

    v16bf af[4], bw[2];
#pragma unroll
    for (int mt = 0; mt < 4; ++mt)
      af[mt] = frag_row(&As[buf][wm * 64 + mt * 16 + l16][0], half);
#pragma unroll
    for (int nt = 0; nt < 2; ++nt)
      bw[nt] = frag_row(&Bs[wn * 32 + nt * 16 + l16][0], half);
#pragma unroll
    for (int mt = 0; mt < 4; ++mt)
#pragma unroll
      for (int nt = 0; nt < 2; ++nt)
        acc[mt][nt] = __builtin_amdgcn_wmma_f32_16x16x32_bf16(
            false, af[mt], false, bw[nt], (short)0, acc[mt][nt], false, false);
    __syncthreads();
  }

  // epilogue: C layout lane holds col n=l16, rows half*8+r (ISA 7.12.2)
#pragma unroll
  for (int mt = 0; mt < 4; ++mt) {
#pragma unroll
    for (int nt = 0; nt < 2; ++nt) {
      const int n = n0 + wn * 32 + nt * 16 + l16;
      if (n >= N) continue;
      const float bv = bias ? bias[n] : 0.f;
#pragma unroll
      for (int r = 0; r < 8; ++r) {
        const int m = m0 + wm * 64 + mt * 16 + half * 8 + r;
        float v = acc[mt][nt][r] + bv;
        if (GELU) {
          const float u = v;
          v = 0.5f * u *
              (1.f + tanhf(0.7978845608028654f * (u + 0.044715f * u * u * u)));
        }
        const size_t idx = (size_t)m * N + n;
        if (RESID)
          Cf[idx] += v;
        else if (OUTBF16)
          Cb[idx] = (bf16)v;
        else
          Cf[idx] = v;
      }
    }
  }
}

// ---------------------------------------------------------------------------
// Flash attention: one workgroup (4 waves) per (b, head, 64-row q-block).
// K block row-major in LDS (S = Q K^T reads it row-wise); V stored TRANSPOSED
// (Vt[d][k]) so P@V B-fragments are contiguous ds_load_b128 too.
// ---------------------------------------------------------------------------
__global__ __launch_bounds__(128) void attn_kernel(const bf16* __restrict__ qkv,
                                                   const float* __restrict__ amask,
                                                   bf16* __restrict__ outp) {
  __shared__ bf16 Qs[64][64];
  __shared__ bf16 Ks[64][64];
  __shared__ bf16 Vt[64][64];  // [d][k]
  __shared__ float Sb[4][16][64];
  __shared__ bf16 Pb[4][16][64];
  __shared__ float Mw[4][16];
  __shared__ float Lw[4][16];
  __shared__ float Aw[4][16];

  const int tid = threadIdx.x;
  const int lane = tid & 31;
  const int w = tid >> 5;
  const int half = lane >> 4;
  const int l16 = lane & 15;
  const int qb = blockIdx.x;  // 0..15 (q block of 64)
  const int hh = blockIdx.y;
  const int b = blockIdx.z;
  const int q0 = qb * 64;
  const size_t rs = 3 * D_MODEL;
  const bf16* qbase = qkv + (size_t)(b * SEQ_T) * rs + hh * HEAD_DIM;
  const bf16* kbase = qbase + D_MODEL;
  const bf16* vbase = qbase + 2 * D_MODEL;

  // stage Q block (64x64 bf16)
#pragma unroll
  for (int i = tid; i < 2048; i += 128) {
    const int r = i >> 5;
    const int c = (i & 31) << 1;
    *(uint32_t*)&Qs[r][c] = *(const uint32_t*)&qbase[(size_t)(q0 + r) * rs + c];
  }
  if (lane < 16) {
    Mw[w][lane] = -3.4028235e38f;
    Lw[w][lane] = 0.f;
  }
  __syncthreads();

  v8f O[4];
#pragma unroll
  for (int dt = 0; dt < 4; ++dt) O[dt] = zero8();

  for (int kb = 0; kb <= qb; ++kb) {
    const int k0 = kb * 64;
#pragma unroll
    for (int i = tid; i < 2048; i += 128) {
      const int r = i >> 5;        // k row
      const int c = (i & 31) << 1; // d pair
      *(uint32_t*)&Ks[r][c] =
          *(const uint32_t*)&kbase[(size_t)(k0 + r) * rs + c];
      const uint32_t u = *(const uint32_t*)&vbase[(size_t)(k0 + r) * rs + c];
      Vt[c + 0][r] = __builtin_bit_cast(bf16, (unsigned short)(u & 0xFFFFu));
      Vt[c + 1][r] = __builtin_bit_cast(bf16, (unsigned short)(u >> 16));
    }
    __syncthreads();

    // S strip [16 q x 64 k] for this wave
#pragma unroll
    for (int kt = 0; kt < 4; ++kt) {
      v8f s = zero8();
#pragma unroll
      for (int d = 0; d < 2; ++d) {
        const v16bf aq = frag_row(&Qs[w * 16 + l16][d * 32], half);
        const v16bf bk = frag_row(&Ks[kt * 16 + l16][d * 32], half);
        s = __builtin_amdgcn_wmma_f32_16x16x32_bf16(false, aq, false, bk,
                                                    (short)0, s, false, false);
      }
      const int kg = k0 + kt * 16 + l16;
      const float add = (1.f - amask[b * SEQ_T + kg]) * (-3.4028235e38f);
#pragma unroll
      for (int r = 0; r < 8; ++r) {
        const int qg = q0 + w * 16 + half * 8 + r;
        float sv = s[r] * 0.125f + add;  // 1/sqrt(64)
        if (kg > qg) sv = -INFINITY;     // causal
        Sb[w][half * 8 + r][kt * 16 + l16] = sv;
      }
    }
    wave_lds_fence();

    // online softmax: 2 lanes per row, 32 cols each
    {
      const int row = lane >> 1, seg = lane & 1;
      const float* srow = &Sb[w][row][seg * 32];
      float mx = -3.4028235e38f;
#pragma unroll
      for (int j = 0; j < 32; ++j) mx = fmaxf(mx, srow[j]);
      mx = fmaxf(mx, __shfl_xor(mx, 1, 32));
      const float m_old = Mw[w][row];
      const float m_new = fmaxf(m_old, mx);
      const float alpha = __expf(m_old - m_new);
      float sum = 0.f;
      bf16* prow = &Pb[w][row][seg * 32];
#pragma unroll
      for (int j = 0; j < 32; ++j) {
        const float p = __expf(srow[j] - m_new);
        prow[j] = (bf16)p;
        sum += p;
      }
      sum += __shfl_xor(sum, 1, 32);
      if (seg == 0) {
        Mw[w][row] = m_new;
        Lw[w][row] = alpha * Lw[w][row] + sum;
        Aw[w][row] = alpha;
      }
    }
    wave_lds_fence();

    // O = alpha*O + P @ V
    {
      float al[8];
#pragma unroll
      for (int r = 0; r < 8; ++r) al[r] = Aw[w][half * 8 + r];
#pragma unroll
      for (int dt = 0; dt < 4; ++dt)
#pragma unroll
        for (int r = 0; r < 8; ++r) O[dt][r] *= al[r];
#pragma unroll
      for (int dt = 0; dt < 4; ++dt) {
#pragma unroll
        for (int ks = 0; ks < 2; ++ks) {
          const v16bf ap = frag_row(&Pb[w][l16][ks * 32], half);
          const v16bf bv = frag_row(&Vt[dt * 16 + l16][ks * 32], half);
          O[dt] = __builtin_amdgcn_wmma_f32_16x16x32_bf16(
              false, ap, false, bv, (short)0, O[dt], false, false);
        }
      }
    }
    __syncthreads();  // K/Vt rewritten next iteration
  }

  float inv[8];
#pragma unroll
  for (int r = 0; r < 8; ++r) inv[r] = 1.f / Lw[w][half * 8 + r];
#pragma unroll
  for (int dt = 0; dt < 4; ++dt) {
#pragma unroll
    for (int r = 0; r < 8; ++r) {
      const int qg = q0 + w * 16 + half * 8 + r;
      outp[(size_t)(b * SEQ_T + qg) * D_MODEL + hh * HEAD_DIM + dt * 16 + l16] =
          (bf16)(O[dt][r] * inv[r]);
    }
  }
}

// ---------------------------------------------------------------------------
// Orchestration
// ---------------------------------------------------------------------------
extern "C" void kernel_launch(void* const* d_in, const int* in_sizes, int n_in,
                              void* d_out, int out_size, void* d_ws,
                              size_t ws_size, hipStream_t stream) {
  (void)in_sizes;
  (void)n_in;
  (void)out_size;
  (void)ws_size;
  const int* x = (const int*)d_in[0];
  const float* amask = (const float*)d_in[1];
  const float* wte = (const float*)d_in[2];
  const float* wpe = (const float*)d_in[3];
  const float* lnf_g = (const float*)d_in[4];
  const float* lnf_b = (const float*)d_in[5];
  const float* lm_head = (const float*)d_in[6];

  char* p = (char*)d_ws;
  float* h = (float*)p;  p += (size_t)ROWS * D_MODEL * 4;
  bf16* xn = (bf16*)p;   p += (size_t)ROWS * D_MODEL * 2;
  bf16* qkvb = (bf16*)p; p += (size_t)ROWS * 3 * D_MODEL * 2;
  bf16* att = (bf16*)p;  p += (size_t)ROWS * D_MODEL * 2;
  bf16* f1 = (bf16*)p;   p += (size_t)ROWS * 4 * D_MODEL * 2;

  embed_kernel<<<ROWS, 256, 0, stream>>>(x, wte, wpe, h);

  for (int l = 0; l < N_LAYERS; ++l) {
    const float* ln1_g = (const float*)d_in[7 + l * 12 + 0];
    const float* ln1_b = (const float*)d_in[7 + l * 12 + 1];
    const float* w_qkv = (const float*)d_in[7 + l * 12 + 2];
    const float* b_qkv = (const float*)d_in[7 + l * 12 + 3];
    const float* w_out = (const float*)d_in[7 + l * 12 + 4];
    const float* b_out = (const float*)d_in[7 + l * 12 + 5];
    const float* ln2_g = (const float*)d_in[7 + l * 12 + 6];
    const float* ln2_b = (const float*)d_in[7 + l * 12 + 7];
    const float* w_fc1 = (const float*)d_in[7 + l * 12 + 8];
    const float* b_fc1 = (const float*)d_in[7 + l * 12 + 9];
    const float* w_fc2 = (const float*)d_in[7 + l * 12 + 10];
    const float* b_fc2 = (const float*)d_in[7 + l * 12 + 11];

    ln_kernel<<<ROWS, 256, 0, stream>>>(h, ln1_g, ln1_b, xn);
    gemm_kernel<0, 0, 1, 0><<<dim3(24, 16), 256, 0, stream>>>(
        xn, w_qkv, b_qkv, nullptr, qkvb, ROWS, 3 * D_MODEL, D_MODEL);
    attn_kernel<<<dim3(SEQ_T / 64, N_HEADS, SEQ_B), 128, 0, stream>>>(
        qkvb, amask, att);
    gemm_kernel<0, 1, 0, 0><<<dim3(8, 16), 256, 0, stream>>>(
        att, w_out, b_out, h, nullptr, ROWS, D_MODEL, D_MODEL);
    ln_kernel<<<ROWS, 256, 0, stream>>>(h, ln2_g, ln2_b, xn);
    gemm_kernel<1, 0, 1, 0><<<dim3(32, 16), 256, 0, stream>>>(
        xn, w_fc1, b_fc1, nullptr, f1, ROWS, 4 * D_MODEL, D_MODEL);
    gemm_kernel<0, 1, 0, 0><<<dim3(8, 16), 256, 0, stream>>>(
        f1, w_fc2, b_fc2, h, nullptr, ROWS, D_MODEL, 4 * D_MODEL);
  }

  ln_kernel<<<ROWS, 256, 0, stream>>>(h, lnf_g, lnf_b, xn);
  gemm_kernel<0, 0, 0, 1><<<dim3((VOCAB + 127) / 128, 16), 256, 0, stream>>>(
      xn, lm_head, nullptr, (float*)d_out, nullptr, ROWS, VOCAB, D_MODEL);
}
